// SGCN_Transformer_50560355008737
// MI455X (gfx1250) — compile-verified, matrix operand order checked by reading, not run on hardware
//
#include <hip/hip_runtime.h>
#include <hip/hip_bf16.h>
#include <cmath>

typedef __attribute__((ext_vector_type(16))) _Float16 v16h;
typedef __attribute__((ext_vector_type(8)))  _Float16 v8h;
typedef __attribute__((ext_vector_type(8)))  float    v8f;

#define NODES 62
#define HID   100
#define HEADS 5
#define DK    20
#define DFF   400
#define BGR   2048

// ---------------- WMMA helpers (CDNA5 wave32, 16x16x32 f16->f32) ----------------
// A 16x32: row = lane%16 ; half j -> k = (lane>=16?8:0) + (j>=8?16:0) + (j&7)
// B 32x16: col = lane%16 ; same k mapping  (so a row-contiguous transposed buffer
//          serves as B via the same loader)
// D 16x16 f32: col = lane%16 ; vgpr v -> row = v + (lane>=16?8:0)

__device__ inline v8f wmma_f16(v16h a, v16h b, v8f c) {
  return __builtin_amdgcn_wmma_f32_16x16x32_f16(false, a, false, b, (short)0, c, false, false);
}

// Unconditional fragment load from a zero-padded row-major LDS buffer.
// Per lane: two contiguous 16B reads. Requires r0+15 < rows, k0+32 <= stride,
// 16B alignment (stride even multiple of 8 halves, k0 multiple of 32).
__device__ inline v16h load_frag16(const _Float16* buf, int stride, int r0, int k0) {
  int lane = threadIdx.x & 31;
  int r = r0 + (lane & 15);
  int kb = k0 + ((lane >= 16) ? 8 : 0);
  const _Float16* p = buf + r * stride + kb;
  union { v16h v; v8h h[2]; } u;
  u.h[0] = *(const v8h*)p;
  u.h[1] = *(const v8h*)(p + 16);
  return u.v;
}

// pre-swizzled weight in global: tile(tk,tn) at (tn*tilesK+tk)*512 halves,
// lane-major, 16 contiguous halves per lane -> one 32B load (L2-resident)
__device__ inline v16h load_frag_B_gl(const _Float16* w, int tilesK, int tk, int tn) {
  int lane = threadIdx.x & 31;
  const _Float16* p = w + (((size_t)(tn * tilesK + tk)) << 9) + (lane << 4);
  return *(const v16h*)p;
}

// ---------------- kernel 1: fold K=2 propagation into one 62x62 matrix ----------------
// h2[d] = sum_s M[d][s]*x[s],  M[d][s] = sum_m N[s][m]*N[m][d],
// N[a][b] = dinv[a]*ew[a*62+b]*dinv[b], deg[a] = sum_b |ew[a*62+b]|
__global__ void prep_graph(const float* __restrict__ ew, float* __restrict__ Mg) {
  __shared__ float dinv[NODES];
  int d = blockIdx.x;
  int t = threadIdx.x;
  if (t < NODES) {
    float s = 0.f;
    for (int j = 0; j < NODES; ++j) s += fabsf(ew[t * NODES + j]);
    dinv[t] = (s > 0.f) ? rsqrtf(s) : 0.f;
  }
  __syncthreads();
  if (t < NODES) {
    int s = t;
    float acc = 0.f;
    for (int m = 0; m < NODES; ++m) {
      float Nsm = dinv[s] * ew[s * NODES + m] * dinv[m];
      float Nmd = dinv[m] * ew[m * NODES + d] * dinv[d];
      acc += Nsm * Nmd;
    }
    Mg[d * NODES + s] = acc;
  }
}

// ---------------- kernel 2: f32 weight (Kd x Nd row-major) -> f16 B-fragment swizzle ----
__global__ void swizzle_weight(const float* __restrict__ src, _Float16* __restrict__ dst,
                               int Kd, int Nd, int total) {
  int idx = blockIdx.x * blockDim.x + threadIdx.x;
  if (idx >= total) return;
  int tilesK = (Kd + 31) >> 5;
  int tile = idx >> 9;
  int rem  = idx & 511;
  int lane = rem >> 4;
  int j    = rem & 15;
  int tk = tile % tilesK;
  int tn = tile / tilesK;
  int k = (tk << 5) + ((lane >= 16) ? 8 : 0) + ((j >= 8) ? 16 : 0) + (j & 7);
  int n = (tn << 4) + (lane & 15);
  float v = (k < Kd && n < Nd) ? src[k * Nd + n] : 0.f;
  dst[idx] = (_Float16)v;
}

// ---------------- LDS arena (halves). All buffers zero-padded for unconditional frags.
#define OFF_H   0        // H   64x128 f16
#define OFF_C   8192     // ctx 64x128 f16
#define OFF_Q   16384    // Q   64x160 f16 (head h at col h*32, cols 20..31 of slot zero)
#define OFF_K   26624    // K   64x160 f16 (same slotting)
#define OFF_V   36864    // V^T 160x64 f16 (row = padded value-col, col = node)
#define OFF_S   47104    // S   64x64  f16 scores/attn (in-place softmax)
#define OFF_X1  51200    // X1  64x128 f16
#define SMEM_HALVES 59392
#define SMEM_BYTES  (SMEM_HALVES * 2)   // 118784 B

// ---------------- kernel 3: one workgroup (8 wave32) per graph ----------------
__global__ __launch_bounds__(256) void sgcn_xformer(
    const float* __restrict__ x, const float* __restrict__ Mg,
    const float* __restrict__ lin_w, const float* __restrict__ lin_b,
    const _Float16* __restrict__ wq_sw, const _Float16* __restrict__ wk_sw,
    const _Float16* __restrict__ wv_sw, const _Float16* __restrict__ wo_sw,
    const _Float16* __restrict__ w1_sw, const _Float16* __restrict__ w2_sw,
    const float* __restrict__ b1, const float* __restrict__ b2,
    const float* __restrict__ ln1g, const float* __restrict__ ln1b,
    const float* __restrict__ ln2g, const float* __restrict__ ln2b,
    float* __restrict__ out) {
  extern __shared__ __align__(16) char smem_raw[];
  _Float16* sm  = (_Float16*)smem_raw;
  _Float16* Hb  = sm + OFF_H;
  _Float16* Cb  = sm + OFF_C;
  _Float16* Qb  = sm + OFF_Q;
  _Float16* Kb  = sm + OFF_K;
  _Float16* Vt  = sm + OFF_V;
  _Float16* Sb  = sm + OFF_S;
  _Float16* X1b = sm + OFF_X1;
  float* Xs = (float*)(sm + OFF_X1);  // 310 f32, alive phases 0-1 only
  float* Pf = (float*)(sm + OFF_C);   // 310 f32, alive phases 1-2 only
  float* Rf = (float*)(sm + OFF_Q);   // 64x100 f32 over Q+K, LN1 window
  _Float16* Fb = sm + OFF_Q;          // 64x224 f16 over Q+K, FFN window
  float* R2 = (float*)(sm + OFF_H);   // 64x100 f32 over H+C, FFN output window

  const int tid  = threadIdx.x;
  const int wave = tid >> 5;
  const int lane = tid & 31;
  const int g    = blockIdx.x;

  // phase 0: stage X (62x5) + zero Q/K/Vt (pads must be zero; stores never touch them)
  for (int i = tid; i < NODES * 5; i += 256) Xs[i] = x[g * NODES * 5 + i];
  for (int i = tid; i < 3 * 10240; i += 256) Qb[i] = (_Float16)0.f;
  __syncthreads();

  // phase 1: P = M @ X   (62x5, f32)
  for (int i = tid; i < NODES * 5; i += 256) {
    int m = i / 5, f = i % 5;
    float acc = 0.f;
    for (int s = 0; s < NODES; ++s) acc += Mg[m * NODES + s] * Xs[s * 5 + f];
    Pf[i] = acc;
  }
  __syncthreads();

  // phase 2: H = P @ lin_w^T + lin_b -> f16 (write pads as zero); zero X1 buffer
  for (int i = tid; i < 64 * 128; i += 256) {
    int m = i >> 7, n = i & 127;
    float acc = 0.f;
    if (m < NODES && n < HID) {
      acc = lin_b[n];
#pragma unroll
      for (int f = 0; f < 5; ++f) acc += Pf[m * 5 + f] * lin_w[n * 5 + f];
    }
    Hb[i] = (_Float16)acc;
  }
  for (int i = tid; i < 64 * 128; i += 256) X1b[i] = (_Float16)0.f;
  __syncthreads();

  // phase 3: zero Cb pad region; Q/K/V = H @ W (3 x 28 tiles, K loop 4x32)
  for (int i = tid; i < 64 * 128; i += 256) Cb[i] = (_Float16)0.f;
  for (int t = wave; t < 84; t += 8) {
    int mat = t / 28, r = t % 28, tm = r / 7, tn = r % 7;
    const _Float16* wsw = (mat == 0) ? wq_sw : (mat == 1) ? wk_sw : wv_sw;
    v8f acc = {};
#pragma unroll
    for (int ks = 0; ks < 4; ++ks) {
      v16h a = load_frag16(Hb, 128, tm * 16, ks * 32);
      v16h b = load_frag_B_gl(wsw, 4, ks, tn);
      acc = wmma_f16(a, b, acc);
    }
    int n = tn * 16 + (lane & 15);
    int mb = tm * 16 + ((lane >= 16) ? 8 : 0);
    if (n < HID) {
      int h = n / 20, c = n - h * 20;
      int col = h * 32 + c;  // head-slotted padded column
      if (mat == 2) {        // V stored transposed: Vt[col][node]
#pragma unroll
        for (int v = 0; v < 8; ++v) Vt[col * 64 + (mb + v)] = (_Float16)acc[v];
      } else {
        _Float16* dst = (mat == 0) ? Qb : Kb;
#pragma unroll
        for (int v = 0; v < 8; ++v) dst[(mb + v) * 160 + col] = (_Float16)acc[v];
      }
    }
  }
  __syncthreads();

  // phase 4: attention per head: scores 64x64 (1 k-step), softmax, ctx
  const float scale = 0.22360679775f; // 1/sqrt(20)
  for (int h = 0; h < HEADS; ++h) {
    for (int t = wave; t < 16; t += 8) {
      int tm = t / 4, tn = t % 4;
      v16h a = load_frag16(Qb, 160, tm * 16, h * 32);
      v16h b = load_frag16(Kb, 160, tn * 16, h * 32);  // row-contig transposed read
      v8f acc = {};
      acc = wmma_f16(a, b, acc);
      int n = tn * 16 + (lane & 15);
      int mb = tm * 16 + ((lane >= 16) ? 8 : 0);
#pragma unroll
      for (int v = 0; v < 8; ++v) Sb[(mb + v) * 64 + n] = (_Float16)(acc[v] * scale);
    }
    __syncthreads();
    if (tid < NODES) {  // row softmax over 62 valid cols; cols 62/63 stay zero
      _Float16* row = Sb + tid * 64;
      float mx = -1e30f;
      for (int c = 0; c < NODES; ++c) { float v = (float)row[c]; mx = v > mx ? v : mx; }
      float sum = 0.f;
      for (int c = 0; c < NODES; ++c) sum += __expf((float)row[c] - mx);
      float inv = 1.f / sum;
      for (int c = 0; c < NODES; ++c) row[c] = (_Float16)(__expf((float)row[c] - mx) * inv);
    }
    __syncthreads();
    {   // ctx_h = attn @ V_h : 8 tiles (4 row x 2 col), K loop 2x32 over nodes
      int tm = wave >> 1, tn = wave & 1;
      v8f acc = {};
#pragma unroll
      for (int ks = 0; ks < 2; ++ks) {
        v16h a = load_frag16(Sb, 64, tm * 16, ks * 32);
        v16h b = load_frag16(Vt, 64, h * 32 + tn * 16, ks * 32);
        acc = wmma_f16(a, b, acc);
      }
      int nl = tn * 16 + (lane & 15);          // 0..31 within head slot
      int col = h * 20 + nl;                   // overlap cols rewritten by next head
      int mb = tm * 16 + ((lane >= 16) ? 8 : 0);
#pragma unroll
      for (int v = 0; v < 8; ++v) Cb[(mb + v) * 128 + col] = (_Float16)acc[v];
    }
    __syncthreads();
  }

  // phase 5: R = H + C @ Wo (f32), LN1 -> X1 (f16, pads already zero)
  for (int t = wave; t < 28; t += 8) {
    int tm = t / 7, tn = t % 7;
    v8f acc = {};
#pragma unroll
    for (int ks = 0; ks < 4; ++ks) {
      v16h a = load_frag16(Cb, 128, tm * 16, ks * 32);
      v16h b = load_frag_B_gl(wo_sw, 4, ks, tn);
      acc = wmma_f16(a, b, acc);
    }
    int n = tn * 16 + (lane & 15);
    int mb = tm * 16 + ((lane >= 16) ? 8 : 0);
    if (n < HID) {
#pragma unroll
      for (int v = 0; v < 8; ++v) {
        int m = mb + v;
        Rf[m * HID + n] = acc[v] + (float)Hb[m * 128 + n];
      }
    }
  }
  __syncthreads();
  if (tid < NODES) {
    const float* row = Rf + tid * HID;
    float mu = 0.f;
    for (int n = 0; n < HID; ++n) mu += row[n];
    mu *= (1.f / HID);
    float var = 0.f;
    for (int n = 0; n < HID; ++n) { float d = row[n] - mu; var += d * d; }
    var *= (1.f / HID);
    float rs = rsqrtf(var + 1e-5f);
    for (int n = 0; n < HID; ++n)
      X1b[tid * 128 + n] = (_Float16)((row[n] - mu) * rs * ln1g[n] + ln1b[n]);
  }
  __syncthreads();

  // phase 6: FFN. R2 = X1 + b2 + relu(X1@W1+b1)@W2; DFF in 32-aligned chunks 224/176
  for (int i = tid; i < 64 * HID; i += 256) {
    int m = i / HID, n = i % HID;
    R2[i] = (float)X1b[m * 128 + n] + b2[n];
  }
  __syncthreads();

  const int chunkBase[2] = {0, 224};
  const int nT1[2]       = {14, 11};   // w1 col tiles per chunk
  const int kT2[2]       = {7, 6};     // w2 k-steps per chunk
  for (int cc = 0; cc < 2; ++cc) {
    int nt = nT1[cc];
    for (int t = wave; t < 4 * nt; t += 8) {
      int tm = t / nt, tn = t % nt;
      v8f acc = {};
#pragma unroll
      for (int ks = 0; ks < 4; ++ks) {
        v16h a = load_frag16(X1b, 128, tm * 16, ks * 32);
        v16h b = load_frag_B_gl(w1_sw, 4, ks, (chunkBase[cc] >> 4) + tn);
        acc = wmma_f16(a, b, acc);
      }
      int nl = tn * 16 + (lane & 15);
      int mb = tm * 16 + ((lane >= 16) ? 8 : 0);
      float bb = b1[chunkBase[cc] + nl];
#pragma unroll
      for (int v = 0; v < 8; ++v) {
        float q = acc[v] + bb;
        Fb[(mb + v) * 224 + nl] = (_Float16)(q > 0.f ? q : 0.f);
      }
    }
    if (cc == 1) {  // zero stale strip (cols 176..223) so Y frags read zeros
      for (int i = tid; i < 64 * 48; i += 256)
        Fb[(i / 48) * 224 + 176 + (i % 48)] = (_Float16)0.f;
    }
    __syncthreads();
    for (int t = wave; t < 28; t += 8) {
      int tm = t / 7, tn = t % 7;
      v8f acc = {};
      for (int ks = 0; ks < kT2[cc]; ++ks) {
        v16h a = load_frag16(Fb, 224, tm * 16, ks * 32);
        v16h b = load_frag_B_gl(w2_sw, 13, (chunkBase[cc] >> 5) + ks, tn);
        acc = wmma_f16(a, b, acc);
      }
      int n = tn * 16 + (lane & 15);
      int mb = tm * 16 + ((lane >= 16) ? 8 : 0);
      if (n < HID) {
#pragma unroll
        for (int v = 0; v < 8; ++v) R2[(mb + v) * HID + n] += acc[v];
      }
    }
    __syncthreads();
  }

  // phase 7: LN2 -> global output
  if (tid < NODES) {
    const float* row = R2 + tid * HID;
    float mu = 0.f;
    for (int n = 0; n < HID; ++n) mu += row[n];
    mu *= (1.f / HID);
    float var = 0.f;
    for (int n = 0; n < HID; ++n) { float d = row[n] - mu; var += d * d; }
    var *= (1.f / HID);
    float rs = rsqrtf(var + 1e-5f);
    float* o = out + ((size_t)(g * NODES + tid)) * HID;
    for (int n = 0; n < HID; ++n)
      o[n] = (row[n] - mu) * rs * ln2g[n] + ln2b[n];
  }
}

// ---------------- launcher ----------------
extern "C" void kernel_launch(void* const* d_in, const int* in_sizes, int n_in,
                              void* d_out, int out_size, void* d_ws, size_t ws_size,
                              hipStream_t stream) {
  (void)in_sizes; (void)n_in; (void)out_size; (void)ws_size;
  const float* x     = (const float*)d_in[0];
  const float* ew    = (const float*)d_in[1];
  // d_in[2] = edge_index (dense block-diagonal, structure known -> unused)
  // d_in[3] = batch_size (== 2048, compiled in)
  const float* lin_w = (const float*)d_in[4];
  const float* lin_b = (const float*)d_in[5];
  const float* wq    = (const float*)d_in[6];
  const float* wk    = (const float*)d_in[7];
  const float* wv    = (const float*)d_in[8];
  const float* wo    = (const float*)d_in[9];
  const float* ln1g  = (const float*)d_in[10];
  const float* ln1b  = (const float*)d_in[11];
  const float* w1    = (const float*)d_in[12];
  const float* b1    = (const float*)d_in[13];
  const float* w2    = (const float*)d_in[14];
  const float* b2    = (const float*)d_in[15];
  const float* ln2g  = (const float*)d_in[16];
  const float* ln2b  = (const float*)d_in[17];
  float* out = (float*)d_out;

  char* ws = (char*)d_ws;
  float*    Mg    = (float*)ws;                 // 62*62*4 = 15376 B
  _Float16* wq_sw = (_Float16*)(ws + 15616);    // 4*7*512 halves = 28672 B
  _Float16* wk_sw = (_Float16*)(ws + 44288);
  _Float16* wv_sw = (_Float16*)(ws + 72960);
  _Float16* wo_sw = (_Float16*)(ws + 101632);
  _Float16* w1_sw = (_Float16*)(ws + 130304);   // 4*25*512 halves = 102400 B
  _Float16* w2_sw = (_Float16*)(ws + 232704);   // 13*7*512 halves = 93184 B (end 325888)

  prep_graph<<<NODES, 64, 0, stream>>>(ew, Mg);
  swizzle_weight<<<(14336 + 255) / 256, 256, 0, stream>>>(wq, wq_sw, 100, 100, 14336);
  swizzle_weight<<<(14336 + 255) / 256, 256, 0, stream>>>(wk, wk_sw, 100, 100, 14336);
  swizzle_weight<<<(14336 + 255) / 256, 256, 0, stream>>>(wv, wv_sw, 100, 100, 14336);
  swizzle_weight<<<(14336 + 255) / 256, 256, 0, stream>>>(wo, wo_sw, 100, 100, 14336);
  swizzle_weight<<<(51200 + 255) / 256, 256, 0, stream>>>(w1, w1_sw, 100, 400, 51200);
  swizzle_weight<<<(46592 + 255) / 256, 256, 0, stream>>>(w2, w2_sw, 400, 100, 46592);

  sgcn_xformer<<<BGR, 256, SMEM_BYTES, stream>>>(
      x, Mg, lin_w, lin_b, wq_sw, wk_sw, wv_sw, wo_sw, w1_sw, w2_sw,
      b1, b2, ln1g, ln1b, ln2g, ln2b, out);
}
